// LocalAttention_549755814221
// MI455X (gfx1250) — matrix operand order repaired and checked by reading, not compiled, not run on hardware
//
#include <hip/hip_runtime.h>
#include <hip/hip_bf16.h>

typedef __attribute__((ext_vector_type(16))) _Float16 v16h;
typedef __attribute__((ext_vector_type(8)))  _Float16 v8h;
typedef __attribute__((ext_vector_type(8)))  float    v8f;

#define NWIN 4096
#define SCALE 0.17677669529663687f   // 32^-0.5
#define NFRAG_W  384   // 48 m-tiles * 8 k-tiles for stacked [Wq;Wkv] (768x256)
#define NFRAG_WO 128   // 16 m-tiles * 8 k-tiles for Wo (256x256)

// ---- LDS layout ----
#define XSTR 264   // halves per X/Out column (256 + pad)
#define YSTR 520   // halves per q/k column (512 + pad)
#define VSTR 72    // halves per v row (64 + pad)
#define DSTR 65    // f32 per dots/attn row (64 + pad, perfect bank spread)
#define ASTR 72    // halves per attn-f16 row (64 + pad, 16B aligned)
#define OFF_Y 33792
#define OFF_V 100352
#define OFF_D 137216
#define OFF_A 153856
#define SMEM_BYTES 163072

__device__ inline v8f wmma16(v16h a, v16h b, v8f c) {
  return __builtin_amdgcn_wmma_f32_16x16x32_f16(false, a, false, b, (short)0, c, false, false);
}

// A-matrix fragment (16x32 f16): lane<16 -> K 0..7 & 16..23 ; lane>=16 -> K 8..15 & 24..31
__device__ inline v16h frag_a(const _Float16* p) {
  int k0 = ((threadIdx.x >> 4) & 1) * 8;
  v8h lo = *(const v8h*)(p + k0);
  v8h hi = *(const v8h*)(p + k0 + 16);
  v16h r;
#pragma unroll
  for (int i = 0; i < 8; ++i) { r[i] = lo[i]; r[i + 8] = hi[i]; }
  return r;
}

// B-matrix fragment (32x16 f16): lane<16 -> K 0..7 & 16..23 ; lane>=16 -> K 8..15 & 24..31
__device__ inline v16h frag_b(const _Float16* p) {
  int k0 = ((threadIdx.x >> 4) & 1) * 8;
  v8h lo = *(const v8h*)(p + k0);
  v8h hi = *(const v8h*)(p + k0 + 16);
  v16h r;
#pragma unroll
  for (int i = 0; i < 8; ++i) { r[i] = lo[i]; r[i + 8] = hi[i]; }
  return r;
}

// ---------------- prep: f32 -> f16 weights in A-fragment-swizzled order + bias gather ----------------
__global__ void __launch_bounds__(256) la_prep(
    const float* __restrict__ Wq, const float* __restrict__ Wkv,
    const float* __restrict__ Wo, const float* __restrict__ pos,
    const int* __restrict__ rel,
    _Float16* __restrict__ w16, _Float16* __restrict__ wo16, float* __restrict__ biasb)
{
  int idx = blockIdx.x * 256 + threadIdx.x;
  const int nW  = NFRAG_W * 512;
  const int nWo = NFRAG_WO * 512;
  if (idx < nW) {
    int frag = idx >> 9, lane = (idx >> 4) & 31, half = idx & 15;
    int mt = frag >> 3, kt = frag & 7;
    int m = mt * 16 + (lane & 15);
    int hl = lane >> 4;
    int kin = (half < 8) ? (hl * 8 + half) : (16 + hl * 8 + (half - 8));
    int k = kt * 32 + kin;
    float v = (m < 256) ? Wq[m * 256 + k] : Wkv[(m - 256) * 256 + k];
    w16[idx] = (_Float16)v;
  } else if (idx < nW + nWo) {
    int j = idx - nW;
    int frag = j >> 9, lane = (j >> 4) & 31, half = j & 15;
    int mt = frag >> 3, kt = frag & 7;
    int m = mt * 16 + (lane & 15);
    int hl = lane >> 4;
    int kin = (half < 8) ? (hl * 8 + half) : (16 + hl * 8 + (half - 8));
    int k = kt * 32 + kin;
    wo16[j] = (_Float16)Wo[m * 256 + k];
  } else if (idx < nW + nWo + 49 * 49) {
    int j = idx - nW - nWo;
    int i = j / 49, jj = j - i * 49;
    int r0 = rel[(i * 49 + jj) * 2 + 0];
    int r1 = rel[(i * 49 + jj) * 2 + 1];
    biasb[j] = pos[r0 * 13 + r1];
  }
}

// ---------------- main: one workgroup (8 wave32) per 7x7 window ----------------
__global__ void __launch_bounds__(256) la_main(
    const float* __restrict__ x, const float* __restrict__ bo,
    const _Float16* __restrict__ w16, const _Float16* __restrict__ wo16,
    const float* __restrict__ biasb,
    float* __restrict__ outImg, float* __restrict__ attnOut)
{
  __shared__ __align__(32) unsigned char smem[SMEM_BYTES];
  _Float16* Xs = (_Float16*)smem;              // X (256x64 col-major), later Out (256x64 col-major)
  _Float16* Ys = (_Float16*)(smem + OFF_Y);    // q,k (512 rows x 64 cols, col-major)
  _Float16* Vs = (_Float16*)(smem + OFF_V);    // v (256 rows x 64 cols, row-major)
  float*    Ds = (float*)(smem + OFF_D);       // dots, then normalized attn, f32 (64 x DSTR)
  _Float16* As = (_Float16*)(smem + OFF_A);    // attn f16 (64 x ASTR)

  const int wg = blockIdx.x;
  const int b  = wg >> 8, rem = wg & 255, wy = rem >> 4, wx = rem & 15;
  const int tid = threadIdx.x;
  const int lane = tid & 31, wv = tid >> 5;
  const int lm = lane & 15, hl = lane >> 4;
  const float* xwin = x + (size_t)b * 256 * 112 * 112 + (size_t)(wy * 7) * 112 + wx * 7;

  // ---- stage X window into LDS (f16, col-major, pad cols 49..63 with zero) ----
  for (int e = tid; e < 256 * 64; e += 256) {
    int c = e >> 6, col = e & 63;
    float v = 0.f;
    if (col < 49) {
      int py = col / 7, px = col - py * 7;
      v = xwin[(size_t)c * (112 * 112) + py * 112 + px];
    }
    Xs[col * XSTR + c] = (_Float16)v;
  }
  __syncthreads();

  // ---- GEMM1: Y(768x64) = [Wq;Wkv](768x256) * X(256x64) ----
  // Each wave owns 6 m-tiles; A fragment reused across the 4 n-tiles (4 live accumulators).
  for (int mtg = 0; mtg < 6; ++mtg) {
    int mt = wv * 6 + mtg;
    v8f acc[4] = {};
#pragma unroll
    for (int kt = 0; kt < 8; ++kt) {
      v16h a = *(const v16h*)(w16 + ((mt * 8 + kt) << 9) + (lane << 4));
#pragma unroll
      for (int nt = 0; nt < 4; ++nt) {
        v16h bb = frag_b(Xs + (nt * 16 + lm) * XSTR + kt * 32);
        acc[nt] = wmma16(a, bb, acc[nt]);
      }
    }
    int mbase = mt * 16 + hl * 8;
    if (mt < 32) {          // q (o<256) and k (256..511): col-major
#pragma unroll
      for (int nt = 0; nt < 4; ++nt) {
        int col = nt * 16 + lm;
#pragma unroll
        for (int r = 0; r < 8; ++r) Ys[col * YSTR + mbase + r] = (_Float16)acc[nt][r];
      }
    } else {                // v (512..767): row-major
#pragma unroll
      for (int nt = 0; nt < 4; ++nt) {
        int col = nt * 16 + lm;
#pragma unroll
        for (int r = 0; r < 8; ++r) Vs[(mbase + r - 512) * VSTR + col] = (_Float16)acc[nt][r];
      }
    }
  }
  __syncthreads();

  // ---- per-head attention ----
  for (int h = 0; h < 8; ++h) {
    // dots(64x64) = q_h(64x32) * k_h^T ; wave: one m-tile, two n-tiles, A reused
    {
      int mt = wv >> 1, ntb = (wv & 1) * 2;
      v16h a = frag_a(Ys + (mt * 16 + lm) * YSTR + h * 32);
#pragma unroll
      for (int ntg = 0; ntg < 2; ++ntg) {
        int nt = ntb + ntg;
        v16h bb = frag_b(Ys + (nt * 16 + lm) * YSTR + 256 + h * 32);
        v8f c = {};
        c = wmma16(a, bb, c);
        int col = nt * 16 + lm, mbase = mt * 16 + hl * 8;
#pragma unroll
        for (int r = 0; r < 8; ++r) Ds[(mbase + r) * DSTR + col] = c[r];
      }
    }
    __syncthreads();

    // softmax: one thread per row; normalized attn kept f32 in Ds, f16 copy in As
    if (tid < 49) {
      float* row = Ds + tid * DSTR;
      _Float16* arow = As + tid * ASTR;
      const float* brow = biasb + tid * 49;
      float mx = -1e30f;
      for (int j = 0; j < 49; ++j) { float d = row[j] * SCALE + brow[j]; mx = fmaxf(mx, d); }
      float s = 0.f;
      for (int j = 0; j < 49; ++j) {
        float e = __expf(row[j] * SCALE + brow[j] - mx);
        s += e;
        row[j] = e;
      }
      float inv = 1.f / s;
      for (int j = 0; j < 64; ++j) {
        float a = (j < 49) ? row[j] * inv : 0.f;
        if (j < 49) row[j] = a;
        arow[j] = (_Float16)a;
      }
    } else if (tid < 64) {   // zero pad rows 49..63 of the f16 attn tile
      _Float16* arow = As + tid * ASTR;
      for (int j = 0; j < 64; ++j) arow[j] = (_Float16)0.f;
    }
    __syncthreads();

    // coalesced cooperative store of attn (f32) to global
    {
      float* ag = attnOut + ((size_t)wg * 8 + h) * 49 * 49;
      for (int e = tid; e < 49 * 49; e += 256) {
        int i = e / 49, j = e - i * 49;
        ag[e] = Ds[i * DSTR + j];
      }
    }

    // out_h(64x32) = attn(64x64) * v_h(64x32) ; K=64 -> 2 WMMA steps; one tile per wave
    {
      int mt = wv >> 1, nt = wv & 1;
      v8f c = {};
#pragma unroll
      for (int kt = 0; kt < 2; ++kt) {
        v16h a  = frag_a(As + (mt * 16 + lm) * ASTR + kt * 32);
        v16h bb = frag_b(Vs + (h * 32 + nt * 16 + lm) * VSTR + kt * 32);
        c = wmma16(a, bb, c);
      }
#pragma unroll
      for (int r = 0; r < 8; ++r) {
        int i = mt * 16 + hl * 8 + r;                 // token (column of Out)
        Xs[i * XSTR + h * 32 + nt * 16 + lm] = (_Float16)c[r];
      }
    }
    __syncthreads();
  }

  // ---- final: Z(256x64) = Wo(256x256) * Out(256x64) + bo ; store to image layout ----
  // Each wave owns 2 m-tiles; A fragment reused across 4 n-tiles.
  for (int mtg = 0; mtg < 2; ++mtg) {
    int mt = wv * 2 + mtg;
    v8f acc[4] = {};
#pragma unroll
    for (int kt = 0; kt < 8; ++kt) {
      v16h a = *(const v16h*)(wo16 + ((mt * 8 + kt) << 9) + (lane << 4));
#pragma unroll
      for (int nt = 0; nt < 4; ++nt) {
        v16h bb = frag_b(Xs + (nt * 16 + lm) * XSTR + kt * 32);
        acc[nt] = wmma16(a, bb, acc[nt]);
      }
    }
#pragma unroll
    for (int nt = 0; nt < 4; ++nt) {
      int col = nt * 16 + lm;
      if (col < 49) {
        int py = col / 7, px = col - py * 7;
        float* op = outImg + (size_t)b * 256 * 112 * 112 + (size_t)(wy * 7 + py) * 112 + wx * 7 + px;
#pragma unroll
        for (int r = 0; r < 8; ++r) {
          int m = mt * 16 + hl * 8 + r;
          op[(size_t)m * (112 * 112)] = acc[nt][r] + bo[m];
        }
      }
    }
  }
}

extern "C" void kernel_launch(void* const* d_in, const int* in_sizes, int n_in,
                              void* d_out, int out_size, void* d_ws, size_t ws_size,
                              hipStream_t stream) {
  (void)in_sizes; (void)n_in; (void)out_size; (void)ws_size;
  const float* x   = (const float*)d_in[0];
  const float* Wq  = (const float*)d_in[1];
  const float* Wkv = (const float*)d_in[2];
  const float* Wo  = (const float*)d_in[3];
  const float* bo  = (const float*)d_in[4];
  const float* pos = (const float*)d_in[5];
  const int*   rel = (const int*)d_in[6];

  _Float16* w16  = (_Float16*)d_ws;
  _Float16* wo16 = (_Float16*)((char*)d_ws + (size_t)NFRAG_W * 512 * 2);
  float*    biasb = (float*)((char*)d_ws + (size_t)(NFRAG_W + NFRAG_WO) * 512 * 2);

  float* outImg  = (float*)d_out;
  float* attnOut = outImg + (size_t)16 * 256 * 112 * 112;

  int prepN = NFRAG_W * 512 + NFRAG_WO * 512 + 49 * 49;
  la_prep<<<(prepN + 255) / 256, 256, 0, stream>>>(Wq, Wkv, Wo, pos, rel, w16, wo16, biasb);
  la_main<<<NWIN, 256, 0, stream>>>(x, bo, w16, wo16, biasb, outImg, attnOut);
}
